// GRUDecoder_50757923504375
// MI455X (gfx1250) — matrix operand order repaired and checked by reading, not compiled, not run on hardware
//
#include <hip/hip_runtime.h>

// ---------------------------------------------------------------------------
// GRU decoder, MI455X (gfx1250, wave32). bf16 WMMA (16x16x32, f32 accum) for
// all big GEMMs; fp32 gates/argmax. Per-step recurrent work fused into one
// kernel per step. 32-row M-tile per block so each W fragment (L2-resident)
// feeds two WMMAs, halving L2 bandwidth demand vs a 16-row tile.
// ---------------------------------------------------------------------------

typedef __attribute__((ext_vector_type(16))) __bf16 v16bf;
typedef __attribute__((ext_vector_type(8)))  __bf16 v8bf;
typedef __attribute__((ext_vector_type(8)))  float  v8f;

#define NN 8192   // B*T token streams
#define HH 512    // hidden
#define EE 128    // emb
#define VV 128    // vocab
#define SS 32     // decode steps
#define G3 1536   // 3*H
#define MR 32     // rows (token streams) per block in the step kernel

// padded LDS row strides (avoid bank conflicts; 64 banks x 4B)
#define HB_LD 520   // bf16 h tile
#define HN_LD 520   // bf16 h_new tile
#define PR_LD 132   // f32 pred tile

__device__ __forceinline__ float sigm_f(float x){ return 1.0f/(1.0f + __expf(-x)); }
__device__ __forceinline__ float tanh_f(float x){ return 1.0f - 2.0f/(__expf(2.0f*x) + 1.0f); }

__device__ __forceinline__ v8f wmma_bf16(v16bf a, v16bf b, v8f c){
  // D = A(16x32 bf16) * B(32x16 bf16) + C(16x16 f32)
  return __builtin_amdgcn_wmma_f32_16x16x32_bf16(false, a, false, b, (short)0, c, false, false);
}

// combine two 8x bf16 16B loads into one 16-element fragment register set
__device__ __forceinline__ v16bf frag16(const __bf16* p0, const __bf16* p1){
  v8bf a = *(const v8bf*)p0;
  v8bf b = *(const v8bf*)p1;
  return __builtin_shufflevector(a, b, 0,1,2,3,4,5,6,7,8,9,10,11,12,13,14,15);
}

__device__ __forceinline__ void zero8(v8f& v){
  #pragma unroll
  for (int e=0;e<8;e++) v[e] = 0.0f;
}

// ---- strided f32 -> row-major bf16 copy (weight repacking / input cast) ----
__global__ void k_conv_bf16(const float* __restrict__ src, int sld, int soff,
                            __bf16* __restrict__ dst, int rows, int cols){
  int i = blockIdx.x*blockDim.x + threadIdx.x;
  if (i >= rows*cols) return;
  int r = i / cols, c = i - r*cols;
  dst[i] = (__bf16)src[(size_t)r*sld + soff + c];
}

__global__ void k_copy_i32(const int* __restrict__ s, int* __restrict__ d, int n){
  int i = blockIdx.x*blockDim.x + threadIdx.x;
  if (i < n) d[i] = s[i];
}

// ---- tiny f32 GEMM: out[r][c] = emb_table[r,:] . W[c, woff:woff+128] -------
__global__ void k_small_gemm(const float* __restrict__ T, const float* __restrict__ W,
                             int wld, int woff, float* __restrict__ outp, int R, int C){
  int i = blockIdx.x*blockDim.x + threadIdx.x;
  if (i >= R*C) return;
  int r = i / C, c = i - r*C;
  const float* t = T + (size_t)r*EE;
  const float* w = W + (size_t)c*wld + woff;
  float s = 0.0f;
  #pragma unroll 4
  for (int k=0;k<EE;k++) s += t[k]*w[k];
  outp[i] = s;
}

// ---- one-time context GEMMs: C[M][Nd] = Abf[M][K] @ Bbf[Nd][K]^T + bias ----
// grid: (Nd/128, M/16), 4 waves per block; each wave owns 2 N-tiles of 16.
__global__ __launch_bounds__(128)
void k_gemm_bf16(const __bf16* __restrict__ A, const __bf16* __restrict__ Bm,
                 const float* __restrict__ bias, float* __restrict__ C,
                 int Nd, int K){
  const int tid  = threadIdx.x;
  const int w    = tid >> 5;
  const int l    = tid & 31;
  const int half = l >> 4;
  const int ln   = l & 15;
  const int rowb = blockIdx.y * 16;
  const int nb   = blockIdx.x * 128 + w * 32;

  v8f a0, a1; zero8(a0); zero8(a1);
  const __bf16* ar  = A  + (size_t)(rowb + ln)*K + half*8;
  const __bf16* b0r = Bm + (size_t)(nb      + ln)*K + half*8;
  const __bf16* b1r = Bm + (size_t)(nb + 16 + ln)*K + half*8;
  for (int k0 = 0; k0 < K; k0 += 32){
    v16bf Af = frag16(ar + k0, ar + k0 + 16);
    a0 = wmma_bf16(Af, frag16(b0r + k0, b0r + k0 + 16), a0);
    a1 = wmma_bf16(Af, frag16(b1r + k0, b1r + k0 + 16), a1);
  }
  #pragma unroll
  for (int v=0; v<8; v++){
    int row = rowb + half*8 + v;
    int c0 = nb + ln, c1 = nb + 16 + ln;
    C[(size_t)row*Nd + c0] = a0[v] + (bias ? bias[c0] : 0.0f);
    C[(size_t)row*Nd + c1] = a1[v] + (bias ? bias[c1] : 0.0f);
  }
}

// ---- fused recurrent step -------------------------------------------------
// Block: 128 threads (4 waves), MR=32 token-rows (two 16-row WMMA tiles that
// share every B fragment).
//   gh = h @ W_hh^T  (bf16 WMMA) -> gates -> h_new
//   pred = h_new @ fc_W_h^T + precomputed(ctx) + vocab-table(emb) -> argmax
__global__ __launch_bounds__(128)
void k_gru_step(int s,
                const float* __restrict__ src_h, float* __restrict__ dst_h,
                int* __restrict__ ch,
                const float* __restrict__ CG,   // [N][1536] ctx@W_ih_ctx^T + b_ih
                const float* __restrict__ GE,   // [V][1536] emb_table@W_ih_emb^T
                const float* __restrict__ bhh,  // [1536]
                const __bf16* __restrict__ Whh, // [1536][512] bf16 row-major
                const __bf16* __restrict__ fWh, // [128][512]  bf16 row-major
                const float* __restrict__ CFC,  // [N][128] ctx@fc_W_ctx^T + fc_b
                const float* __restrict__ FE,   // [V][128] emb_table@fc_W_emb^T
                float* __restrict__ outp){
  __shared__ __bf16 hb[MR*HB_LD];
  __shared__ __bf16 hn[MR*HN_LD];
  __shared__ float  pr[MR*PR_LD];
  __shared__ int    chl[MR];

  const int tid  = threadIdx.x;
  const int w    = tid >> 5;
  const int l    = tid & 31;
  const int half = l >> 4;
  const int ln   = l & 15;
  const int rowb = blockIdx.x * MR;

  // Phase 0: stage h tile (32x512) into LDS as bf16; fetch chars.
  #pragma unroll
  for (int i = 0; i < 32; ++i){
    int f4 = tid + i*128;                 // 4096 float4 chunks
    int r  = f4 >> 7;
    int c  = (f4 & 127) * 4;
    const float4 v = *(const float4*)(src_h + (size_t)(rowb + r)*HH + c);
    __bf16* d = &hb[r*HB_LD + c];
    d[0]=(__bf16)v.x; d[1]=(__bf16)v.y; d[2]=(__bf16)v.z; d[3]=(__bf16)v.w;
  }
  if (tid < MR) chl[tid] = ch[rowb + tid];
  __syncthreads();

  // Phase 1: per wave, 4 chunks of 32 H-columns. For each chunk compute the
  // r/z/n gh tiles for BOTH 16-row tiles (12 accumulators; B loaded once).
  const __bf16* ap0 = &hb[(     ln)*HB_LD + half*8];
  const __bf16* ap1 = &hb[(16 + ln)*HB_LD + half*8];
  #pragma unroll 1
  for (int chunk = 0; chunk < 4; ++chunk){
    const int c0 = w*128 + chunk*32;
    const int cb[6] = {c0, c0+16, c0+512, c0+528, c0+1024, c0+1040};
    v8f acc[2][6];
    #pragma unroll
    for (int rt=0;rt<2;rt++)
      #pragma unroll
      for (int i=0;i<6;i++) zero8(acc[rt][i]);
    const __bf16* bp[6];
    #pragma unroll
    for (int i=0;i<6;i++) bp[i] = Whh + (size_t)(cb[i] + ln)*HH + half*8;

    #pragma unroll 1
    for (int k0 = 0; k0 < HH; k0 += 32){
      v16bf A0 = frag16(ap0 + k0, ap0 + k0 + 16);
      v16bf A1 = frag16(ap1 + k0, ap1 + k0 + 16);
      #pragma unroll
      for (int i=0;i<6;i++){
        v16bf Bf = frag16(bp[i] + k0, bp[i] + k0 + 16);
        acc[0][i] = wmma_bf16(A0, Bf, acc[0][i]);
        acc[1][i] = wmma_bf16(A1, Bf, acc[1][i]);
      }
    }

    #pragma unroll
    for (int rt=0; rt<2; rt++){
      #pragma unroll
      for (int t=0; t<2; t++){
        const int c = c0 + t*16 + ln;                 // H column 0..511
        const float bh_r = bhh[c], bh_z = bhh[c+512], bh_n = bhh[c+1024];
        #pragma unroll
        for (int v=0; v<8; v++){
          const int lrow = rt*16 + half*8 + v;
          const int row  = rowb + lrow;
          const int cc   = chl[lrow];
          const size_t gb = (size_t)row*G3 + c;
          const size_t eb = (size_t)cc*G3 + c;
          float rr = sigm_f(CG[gb]      + GE[eb]      + acc[rt][  t][v] + bh_r);
          float zz = sigm_f(CG[gb+512]  + GE[eb+512]  + acc[rt][2+t][v] + bh_z);
          float nn = tanh_f(CG[gb+1024] + GE[eb+1024] + rr*(acc[rt][4+t][v] + bh_n));
          float hv = src_h[(size_t)row*HH + c];
          float h2 = (1.0f - zz)*nn + zz*hv;
          dst_h[(size_t)row*HH + c] = h2;
          hn[lrow*HN_LD + c] = (__bf16)h2;
        }
      }
    }
  }
  __syncthreads();

  // Phase 2: pred tile = h_new @ fc_W_h^T (each wave: 32 vocab cols, both
  // row tiles; B fragments shared).
  {
    v8f p00, p01, p10, p11;
    zero8(p00); zero8(p01); zero8(p10); zero8(p11);
    const __bf16* an0 = &hn[(     ln)*HN_LD + half*8];
    const __bf16* an1 = &hn[(16 + ln)*HN_LD + half*8];
    const __bf16* b0 = fWh + (size_t)(w*32      + ln)*HH + half*8;
    const __bf16* b1 = fWh + (size_t)(w*32 + 16 + ln)*HH + half*8;
    #pragma unroll 1
    for (int k0 = 0; k0 < HH; k0 += 32){
      v16bf A0 = frag16(an0 + k0, an0 + k0 + 16);
      v16bf A1 = frag16(an1 + k0, an1 + k0 + 16);
      v16bf B0 = frag16(b0 + k0, b0 + k0 + 16);
      v16bf B1 = frag16(b1 + k0, b1 + k0 + 16);
      p00 = wmma_bf16(A0, B0, p00);
      p01 = wmma_bf16(A0, B1, p01);
      p10 = wmma_bf16(A1, B0, p10);
      p11 = wmma_bf16(A1, B1, p11);
    }
    #pragma unroll
    for (int rt=0; rt<2; rt++){
      const v8f& q0 = rt ? p10 : p00;
      const v8f& q1 = rt ? p11 : p01;
      #pragma unroll
      for (int v=0; v<8; v++){
        const int lrow = rt*16 + half*8 + v;
        const int row  = rowb + lrow;
        const int cc   = chl[lrow];
        int c = w*32 + ln;
        float pa = q0[v] + CFC[(size_t)row*VV + c] + FE[cc*VV + c];
        outp[((size_t)row*SS + s)*VV + c] = pa;
        pr[lrow*PR_LD + c] = pa;
        c += 16;
        float pb = q1[v] + CFC[(size_t)row*VV + c] + FE[cc*VV + c];
        outp[((size_t)row*SS + s)*VV + c] = pb;
        pr[lrow*PR_LD + c] = pb;
      }
    }
  }
  __syncthreads();

  // Phase 3: argmax over 128 vocab entries per row (first-max semantics).
  if (tid < MR){
    const float* p = &pr[tid*PR_LD];
    float best = p[0]; int bi = 0;
    for (int k=1;k<VV;k++){ float pv = p[k]; if (pv > best){ best = pv; bi = k; } }
    ch[rowb + tid] = bi;
  }
}

// ---------------------------------------------------------------------------
extern "C" void kernel_launch(void* const* d_in, const int* in_sizes, int n_in,
                              void* d_out, int out_size, void* d_ws, size_t ws_size,
                              hipStream_t stream){
  const float* encoded   = (const float*)d_in[0];  // [B,T,H] == [N,H]
  const int*   init_char = (const int*)  d_in[1];  // [N]
  const float* emb_table = (const float*)d_in[3];  // [V,E]
  const float* W_ih      = (const float*)d_in[4];  // [3H, E+H]
  const float* W_hh      = (const float*)d_in[5];  // [3H, H]
  const float* b_ih      = (const float*)d_in[6];  // [3H]
  const float* b_hh      = (const float*)d_in[7];  // [3H]
  const float* fc_W      = (const float*)d_in[8];  // [V, E+2H]
  const float* fc_b      = (const float*)d_in[9];  // [V]
  float* out = (float*)d_out;

  // workspace carve-up (~101 MB), 256B aligned
  char* ws = (char*)d_ws;
  size_t off = 0;
  auto take = [&](size_t bytes)->char*{
    char* p = ws + off; off = (off + bytes + 255) & ~(size_t)255; return p; };
  float*  CG   = (float*) take((size_t)NN*G3*4);   // ctx gate term + b_ih
  float*  CFC  = (float*) take((size_t)NN*VV*4);   // ctx fc term + fc_b
  float*  GE   = (float*) take((size_t)VV*G3*4);   // vocab gate table
  float*  FE   = (float*) take((size_t)VV*VV*4);   // vocab fc table
  __bf16* Whhb = (__bf16*)take((size_t)G3*HH*2);
  __bf16* Wicb = (__bf16*)take((size_t)G3*HH*2);   // W_ih ctx part
  __bf16* Wfcc = (__bf16*)take((size_t)VV*HH*2);   // fc_W ctx part
  __bf16* Wfch = (__bf16*)take((size_t)VV*HH*2);   // fc_W h part
  __bf16* Cbf  = (__bf16*)take((size_t)NN*HH*2);   // context bf16
  float*  h0   = (float*) take((size_t)NN*HH*4);
  float*  h1   = (float*) take((size_t)NN*HH*4);
  int*    chb  = (int*)   take((size_t)NN*4);

  auto cdiv = [](long a, long b)->int{ return (int)((a + b - 1)/b); };

  // 1) weight / input repacks to bf16
  k_conv_bf16<<<cdiv((long)G3*HH,256),256,0,stream>>>(W_hh, 512,   0, Whhb, G3, HH);
  k_conv_bf16<<<cdiv((long)G3*HH,256),256,0,stream>>>(W_ih, 640, 128, Wicb, G3, HH);
  k_conv_bf16<<<cdiv((long)VV*HH,256),256,0,stream>>>(fc_W,1152, 128, Wfcc, VV, HH);
  k_conv_bf16<<<cdiv((long)VV*HH,256),256,0,stream>>>(fc_W,1152, 640, Wfch, VV, HH);
  k_conv_bf16<<<cdiv((long)NN*HH,256),256,0,stream>>>(encoded,512, 0, Cbf,  NN, HH);
  k_copy_i32 <<<cdiv(NN,256),256,0,stream>>>(init_char, chb, NN);

  // 2) vocab tables (exact f32)
  k_small_gemm<<<cdiv((long)VV*G3,256),256,0,stream>>>(emb_table, W_ih,  640, 0, GE, VV, G3);
  k_small_gemm<<<cdiv((long)VV*VV,256),256,0,stream>>>(emb_table, fc_W, 1152, 0, FE, VV, VV);

  // 3) one-time context GEMMs (bf16 WMMA)
  { dim3 g(G3/128, NN/16); k_gemm_bf16<<<g,128,0,stream>>>(Cbf, Wicb, b_ih, CG, G3, HH); }
  { dim3 g(VV/128, NN/16); k_gemm_bf16<<<g,128,0,stream>>>(Cbf, Wfcc, fc_b, CFC, VV, HH); }

  // 4) 32 sequential fused GRU steps (h ping-pongs; step 0 reads encoded)
  for (int s = 0; s < SS; ++s){
    const float* src = (s == 0) ? encoded : ((s & 1) ? h0 : h1);
    float*       dst = (s & 1) ? h1 : h0;
    k_gru_step<<<NN/MR, 128, 0, stream>>>(s, src, dst, chb,
                                          CG, GE, b_hh, Whhb, Wfch, CFC, FE, out);
  }
  (void)in_sizes; (void)n_in; (void)out_size; (void)ws_size;
}
